// Mamba_53687091200064
// MI455X (gfx1250) — compile-verified, matrix operand order checked by reading, not run on hardware
//
#include <hip/hip_runtime.h>
#include <hip/hip_bf16.h>
#include <cstdint>
#include <cstddef>

// ---------------- problem dimensions (fixed by the reference) ----------------
#define BB 2
#define LL 512
#define HH 2048
#define DI 4096
#define NN 16
#define KK 4
#define RR 128
#define BL (BB*LL)        // 1024 rows
#define SPW (RR + 2*NN)   // 160

typedef __bf16 bf16_t;
typedef __bf16 v16bf __attribute__((ext_vector_type(16)));
typedef float  v8f   __attribute__((ext_vector_type(8)));

// ---------------- elementwise helpers ----------------
__global__ void cvt_f32_bf16(const float* __restrict__ s, bf16_t* __restrict__ d, int n) {
  int i = blockIdx.x * blockDim.x + threadIdx.x;
  if (i < n) d[i] = (bf16_t)s[i];
}

__global__ void cvt_rows_f32_bf16(const float* __restrict__ s, bf16_t* __restrict__ d,
                                  int rows, int src_ld, int cols) {
  int i = blockIdx.x * blockDim.x + threadIdx.x;
  if (i >= rows * cols) return;
  int r = i / cols, c = i % cols;
  d[i] = (bf16_t)s[(size_t)r * src_ld + c];
}

__global__ void softplus_inplace(float* __restrict__ p, int n) {
  int i = blockIdx.x * blockDim.x + threadIdx.x;
  if (i < n) {
    float x = p[i];
    p[i] = (x > 20.f) ? x : log1pf(__expf(x));
  }
}

// ---------------- depthwise causal conv (K=4) + SiLU ----------------
// proj: (BL, 2*DI); h = proj[:, :DI].  Writes hs (f32) and hs_bf (bf16).
__global__ void conv_silu(const float* __restrict__ proj,
                          const float* __restrict__ conv_w,  // (DI, 1, KK)
                          const float* __restrict__ conv_b,  // (DI)
                          float* __restrict__ hs,
                          bf16_t* __restrict__ hs_bf) {
  int i = blockIdx.x * blockDim.x + threadIdx.x;  // over BL*DI
  if (i >= BL * DI) return;
  int d  = i % DI;
  int bl = i / DI;
  int l  = bl % LL;
  int b  = bl / LL;
  float acc = conv_b[d];
#pragma unroll
  for (int k = 0; k < KK; ++k) {
    int ls = l - (KK - 1) + k;
    if (ls >= 0)
      acc += proj[((size_t)(b * LL + ls)) * (2 * DI) + d] * conv_w[d * KK + k];
  }
  float sv = acc / (1.f + __expf(-acc));   // SiLU
  hs[i] = sv;
  hs_bf[i] = (bf16_t)sv;
}

// ---------------- WMMA bf16 GEMM:  C(MxN) = A(MxK) * B(NxK)^T + bias -----------
// Double-buffered LDS staging via GLOBAL_LOAD_ASYNC_TO_LDS_B128 (ASYNCcnt),
// overlapped with the 8-WMMA compute block of the current tile.
#define TM 128
#define TN 128
#define TK 32

__global__ __launch_bounds__(256) void gemm_bf16_wmma(
    const bf16_t* __restrict__ A,    // M x Kd row-major
    const bf16_t* __restrict__ Bw,   // N x Kd row-major (weights)
    const float*  __restrict__ bias, // N or nullptr
    float* __restrict__ C,           // M x N row-major
    int M, int N, int Kd) {
  __shared__ bf16_t sA[2][TM][TK + 8];
  __shared__ bf16_t sB[2][TN][TK + 8];

  const int tid  = threadIdx.x;
  const int lane = tid & 31;
  const int wave = tid >> 5;       // 0..7
  const int wr   = wave >> 2;      // 0..1  -> 64-row band
  const int wc   = wave & 3;       // 0..3  -> 32-col band
  const int hfid = lane >> 4;      // lane half (0/1)
  const int lr   = lane & 15;

  const int m0 = blockIdx.y * TM;
  const int n0 = blockIdx.x * TN;

  const v8f vzero = {0.f, 0.f, 0.f, 0.f, 0.f, 0.f, 0.f, 0.f};
  v8f acc[4][2];
#pragma unroll
  for (int i = 0; i < 4; ++i)
#pragma unroll
    for (int j = 0; j < 2; ++j) acc[i][j] = vzero;

  union FragU { v16bf v; uint4 u[2]; };

  // stage one 128x32 K-tile of A and B directly into LDS (async, no VGPR hop).
  // B rows with n >= N are left unwritten: they only feed output columns >= N,
  // which are never stored.
  auto stage = [&](int kk, int buf) {
#pragma unroll
    for (int it = 0; it < 2; ++it) {
      int idx = tid + it * 256;    // 0..511
      int row = idx >> 2;          // 0..127
      int seg = idx & 3;           // 16B segment of the 64B row
      int gr = m0 + row;
      if (gr < M) {
        uint32_t loff = (uint32_t)(uintptr_t)&sA[buf][row][seg * 8];
        const bf16_t* gp = A + (size_t)gr * Kd + kk + seg * 8;
        asm volatile("global_load_async_to_lds_b128 %0, %1, off"
                     :: "v"(loff), "v"(gp) : "memory");
      }
      int gn = n0 + row;
      if (gn < N) {
        uint32_t loff = (uint32_t)(uintptr_t)&sB[buf][row][seg * 8];
        const bf16_t* gp = Bw + (size_t)gn * Kd + kk + seg * 8;
        asm volatile("global_load_async_to_lds_b128 %0, %1, off"
                     :: "v"(loff), "v"(gp) : "memory");
      }
    }
  };

  const int nk = Kd / TK;          // Kd is a multiple of 32 for all call sites
  stage(0, 0);
  asm volatile("s_wait_asynccnt 0" ::: "memory");
  __syncthreads();

  for (int ki = 0; ki < nk; ++ki) {
    const int buf = ki & 1;
    if (ki + 1 < nk) stage((ki + 1) * TK, buf ^ 1);  // overlap with compute

    // load fragments per ISA 16-bit A/B layouts
    FragU af[4], bfr[2];
#pragma unroll
    for (int i = 0; i < 4; ++i) {
      const bf16_t* p = &sA[buf][wr * 64 + i * 16 + lr][0];
      af[i].u[0] = *(const uint4*)(p + hfid * 8);
      af[i].u[1] = *(const uint4*)(p + 16 + hfid * 8);
    }
#pragma unroll
    for (int j = 0; j < 2; ++j) {
      const bf16_t* p = &sB[buf][wc * 32 + j * 16 + lr][0];
      bfr[j].u[0] = *(const uint4*)(p + hfid * 8);
      bfr[j].u[1] = *(const uint4*)(p + 16 + hfid * 8);
    }

#pragma unroll
    for (int i = 0; i < 4; ++i)
#pragma unroll
      for (int j = 0; j < 2; ++j)
        acc[i][j] = __builtin_amdgcn_wmma_f32_16x16x32_bf16(
            false, af[i].v, false, bfr[j].v, (short)0, acc[i][j], false, false);

    // next tile's async copies must have landed before anyone reads buf^1;
    // barrier also protects buf from being overwritten while still being read.
    asm volatile("s_wait_asynccnt 0" ::: "memory");
    __syncthreads();
  }

  // store C (f32 16x16 D layout: VGPR e -> row hfid*8+e, lane lr -> col)
#pragma unroll
  for (int i = 0; i < 4; ++i) {
#pragma unroll
    for (int j = 0; j < 2; ++j) {
      int col = n0 + wc * 32 + j * 16 + lr;
      if (col < N) {
        float bv = bias ? bias[col] : 0.f;
#pragma unroll
        for (int e = 0; e < 8; ++e) {
          int row = m0 + wr * 64 + i * 16 + hfid * 8 + e;
          if (row < M) C[(size_t)row * N + col] = acc[i][j][e] + bv;
        }
      }
    }
  }
}

// ---------------- fused selective scan + epilogue ----------------
// one lane per (b, d); N=16 states in registers; writes y_bf for out-proj GEMM
__global__ __launch_bounds__(256) void scan_kernel(
    const float* __restrict__ dt,    // (BL, DI)  (post-softplus)
    const float* __restrict__ hs,    // (BL, DI)
    const float* __restrict__ sp,    // (BL, SPW): [.. ts(128) | Bm(16) | Cm(16)]
    const float* __restrict__ proj,  // (BL, 2*DI): gate at +DI
    const float* __restrict__ A_log, // (DI, NN)
    const float* __restrict__ Dvec,  // (DI)
    bf16_t* __restrict__ y_bf) {     // (BL, DI)
  int idx = blockIdx.x * blockDim.x + threadIdx.x;  // over BB*DI
  if (idx >= BB * DI) return;
  int d = idx % DI;
  int b = idx / DI;

  float An[NN];
#pragma unroll
  for (int n = 0; n < NN; ++n) An[n] = -__expf(A_log[d * NN + n]);
  float Dd = Dvec[d];

  float s[NN];
#pragma unroll
  for (int n = 0; n < NN; ++n) s[n] = 0.f;

  for (int l = 0; l < LL; ++l) {
    size_t r = (size_t)(b * LL + l);
    float dtv = dt[r * DI + d];
    float hv  = hs[r * DI + d];
    float g   = proj[r * (2 * DI) + DI + d];

    const float4* bp = (const float4*)(sp + r * SPW + RR);
    float Bn[NN], Cn[NN];
    ((float4*)Bn)[0] = bp[0]; ((float4*)Bn)[1] = bp[1];
    ((float4*)Bn)[2] = bp[2]; ((float4*)Bn)[3] = bp[3];
    ((float4*)Cn)[0] = bp[4]; ((float4*)Cn)[1] = bp[5];
    ((float4*)Cn)[2] = bp[6]; ((float4*)Cn)[3] = bp[7];

    float dtu = dtv * hv;
    float y = 0.f;
#pragma unroll
    for (int n = 0; n < NN; ++n) {
      float dA = __expf(dtv * An[n]);
      s[n] = dA * s[n] + dtu * Bn[n];
      y += s[n] * Cn[n];
    }
    float sg = g / (1.f + __expf(-g));   // silu(gate)
    y = (y + hv * Dd) * sg;
    y_bf[r * DI + d] = (bf16_t)y;
  }
}

// ---------------- host-side launcher ----------------
extern "C" void kernel_launch(void* const* d_in, const int* in_sizes, int n_in,
                              void* d_out, int out_size, void* d_ws, size_t ws_size,
                              hipStream_t stream) {
  (void)in_sizes; (void)n_in; (void)out_size; (void)ws_size;
  const float* x       = (const float*)d_in[0];
  const float* in_w    = (const float*)d_in[1];
  const float* in_b    = (const float*)d_in[2];
  const float* conv_w  = (const float*)d_in[3];
  const float* conv_b  = (const float*)d_in[4];
  const float* xproj_w = (const float*)d_in[5];
  const float* dt_w    = (const float*)d_in[6];
  const float* dt_b    = (const float*)d_in[7];
  const float* A_log   = (const float*)d_in[8];
  const float* Dvec    = (const float*)d_in[9];
  const float* out_w   = (const float*)d_in[10];
  const float* out_b   = (const float*)d_in[11];
  float* out = (float*)d_out;

  char* wsb = (char*)d_ws;
  size_t off = 0;
  auto alloc = [&](size_t bytes) -> char* {
    char* p = wsb + off;
    off = (off + bytes + 255) & ~(size_t)255;
    return p;
  };
  bf16_t* x_bf    = (bf16_t*)alloc((size_t)BL * HH * 2);
  bf16_t* in_w_bf = (bf16_t*)alloc((size_t)2 * DI * HH * 2);
  bf16_t* xp_bf   = (bf16_t*)alloc((size_t)SPW * DI * 2);
  bf16_t* dtw_bf  = (bf16_t*)alloc((size_t)DI * RR * 2);
  bf16_t* ow_bf   = (bf16_t*)alloc((size_t)HH * DI * 2);
  float*  proj    = (float*) alloc((size_t)BL * 2 * DI * 4);
  float*  hs      = (float*) alloc((size_t)BL * DI * 4);
  bf16_t* hs_bf   = (bf16_t*)alloc((size_t)BL * DI * 2);
  float*  sp      = (float*) alloc((size_t)BL * SPW * 4);
  bf16_t* ts_bf   = (bf16_t*)alloc((size_t)BL * RR * 2);
  float*  dtbuf   = (float*) alloc((size_t)BL * DI * 4);
  bf16_t* y_bf    = (bf16_t*)alloc((size_t)BL * DI * 2);

  auto cdiv = [](int a, int b) { return (a + b - 1) / b; };

  // fp32 -> bf16 conversions (weights + activations entering WMMA)
  cvt_f32_bf16<<<cdiv(BL * HH, 256), 256, 0, stream>>>(x, x_bf, BL * HH);
  cvt_f32_bf16<<<cdiv(2 * DI * HH, 256), 256, 0, stream>>>(in_w, in_w_bf, 2 * DI * HH);
  cvt_f32_bf16<<<cdiv(SPW * DI, 256), 256, 0, stream>>>(xproj_w, xp_bf, SPW * DI);
  cvt_f32_bf16<<<cdiv(DI * RR, 256), 256, 0, stream>>>(dt_w, dtw_bf, DI * RR);
  cvt_f32_bf16<<<cdiv(HH * DI, 256), 256, 0, stream>>>(out_w, ow_bf, HH * DI);

  // proj = x @ in_w^T + in_b     (1024 x 8192, K=2048)
  {
    dim3 g(cdiv(2 * DI, TN), cdiv(BL, TM));
    gemm_bf16_wmma<<<g, 256, 0, stream>>>(x_bf, in_w_bf, in_b, proj, BL, 2 * DI, HH);
  }

  // depthwise conv + SiLU -> hs, hs_bf
  conv_silu<<<cdiv(BL * DI, 256), 256, 0, stream>>>(proj, conv_w, conv_b, hs, hs_bf);

  // sp = hs @ xproj_w^T          (1024 x 160, K=4096)
  {
    dim3 g(cdiv(SPW, TN), cdiv(BL, TM));
    gemm_bf16_wmma<<<g, 256, 0, stream>>>(hs_bf, xp_bf, nullptr, sp, BL, SPW, DI);
  }

  // ts (first 128 cols of sp) -> bf16
  cvt_rows_f32_bf16<<<cdiv(BL * RR, 256), 256, 0, stream>>>(sp, ts_bf, BL, SPW, RR);

  // dtraw = ts @ dt_w^T + dt_b   (1024 x 4096, K=128)
  {
    dim3 g(cdiv(DI, TN), cdiv(BL, TM));
    gemm_bf16_wmma<<<g, 256, 0, stream>>>(ts_bf, dtw_bf, dt_b, dtbuf, BL, DI, RR);
  }

  // dt = softplus(dtraw)
  softplus_inplace<<<cdiv(BL * DI, 256), 256, 0, stream>>>(dtbuf, BL * DI);

  // fused selective scan + (y + hs*D)*silu(gate) -> y_bf
  scan_kernel<<<cdiv(BB * DI, 256), 256, 0, stream>>>(dtbuf, hs, sp, proj, A_log, Dvec, y_bf);

  // out = y @ out_w^T + out_b    (1024 x 2048, K=4096) -> d_out (f32)
  {
    dim3 g(cdiv(HH, TN), cdiv(BL, TM));
    gemm_bf16_wmma<<<g, 256, 0, stream>>>(y_bf, ow_bf, out_b, out, BL, HH, DI);
  }
}
